// GGNNActor_8916352106913
// MI455X (gfx1250) — compile-verified, matrix operand order checked by reading, not run on hardware
//
#include <hip/hip_runtime.h>

#define N_NODES_C 50000
#define N_EDGES_C 800000
#define F_IN_C 32
#define C_C 256
#define N_PAD 50016          // multiple of 32
#define N_ACT_C 8

typedef __attribute__((ext_vector_type(16))) __bf16 v16bf;
typedef __attribute__((ext_vector_type(8)))  __bf16 v8bf;
typedef __attribute__((ext_vector_type(8)))  float  v8f;

static __device__ __forceinline__ v8f wmma_bf16(v16bf a, v16bf b, v8f c) {
    // D = A(16x32 bf16) x B(32x16 bf16) + C(16x16 f32)
    return __builtin_amdgcn_wmma_f32_16x16x32_bf16(
        /*neg_a=*/false, a, /*neg_b=*/false, b,
        /*c_mod=*/(short)0, c, /*reuse_a=*/false, /*reuse_b=*/false);
}

// Build 16x32 A fragment from a row-major f32 matrix (ld floats per row),
// converting to bf16 in-register.  Per ISA 7.12.2 (16-bit A 16x32):
//   lane<16  holds row M=lane,     K = k0+0..7  then k0+16..23
//   lane>=16 holds row M=lane-16,  K = k0+8..15 then k0+24..31
static __device__ __forceinline__ v16bf load_a_frag_f32(
    const float* __restrict__ A, int row, int ld, int k0, int lane) {
    int half = lane >> 4;
    const float* p0 = A + (long long)row * ld + k0 + half * 8;
    const float* p1 = p0 + 16;
    float4 f0 = *(const float4*)(p0);
    float4 f1 = *(const float4*)(p0 + 4);
    float4 f2 = *(const float4*)(p1);
    float4 f3 = *(const float4*)(p1 + 4);
    v16bf a;
    a[0]  = (__bf16)f0.x; a[1]  = (__bf16)f0.y; a[2]  = (__bf16)f0.z; a[3]  = (__bf16)f0.w;
    a[4]  = (__bf16)f1.x; a[5]  = (__bf16)f1.y; a[6]  = (__bf16)f1.z; a[7]  = (__bf16)f1.w;
    a[8]  = (__bf16)f2.x; a[9]  = (__bf16)f2.y; a[10] = (__bf16)f2.z; a[11] = (__bf16)f2.w;
    a[12] = (__bf16)f3.x; a[13] = (__bf16)f3.y; a[14] = (__bf16)f3.z; a[15] = (__bf16)f3.w;
    return a;
}

// Build 32x16 B fragment from a bf16 matrix stored as BT[outCol][k] (ld = K
// stride).  Mirrors A layout with M->N: lane holds column n = lane&15, K
// ranges split by lane-half.  16-byte contiguous loads in K.
static __device__ __forceinline__ v16bf load_b_frag_bf16(
    const __bf16* __restrict__ BT, int col, int ld, int k0, int lane) {
    int half = lane >> 4;
    const __bf16* p = BT + (long long)col * ld + k0 + half * 8;
    v8bf b0 = *(const v8bf*)(p);
    v8bf b1 = *(const v8bf*)(p + 16);
    v16bf b;
#pragma unroll
    for (int i = 0; i < 8; ++i) { b[i] = b0[i]; b[8 + i] = b1[i]; }
    return b;
}

static __device__ __forceinline__ float sigmoidf_(float x) {
    return 1.0f / (1.0f + __expf(-x));
}

// ---------------------------------------------------------------------------
// One-time weight conversion to bf16.  ggT[l][j][k] = gg[l][k][j] (transpose
// so B fragments are K-contiguous); w_ih/w_hh ([3C,C], torch layout) are
// already [outCol][k] for gi = m @ w_ih^T, so plain cast.
__global__ void convw_kernel(const float* __restrict__ gg,
                             const float* __restrict__ wih,
                             const float* __restrict__ whh,
                             __bf16* __restrict__ ggT,
                             __bf16* __restrict__ wihb,
                             __bf16* __restrict__ whhb) {
    int i = blockIdx.x * 256 + threadIdx.x;
    if (i < 3 * 256 * 256) {
        int l = i >> 16, r = i & 65535, j = r >> 8, k = r & 255;
        ggT[i] = (__bf16)gg[l * 65536 + k * 256 + j];
    }
    if (i < 768 * 256) {
        wihb[i] = (__bf16)wih[i];
        whhb[i] = (__bf16)whh[i];
    }
}

// h[n][j] = x[n][j] for j<32, else 0; pad rows fully zero.
__global__ void init_h_kernel(const float* __restrict__ x, float* __restrict__ h) {
    int i = blockIdx.x * 256 + threadIdx.x;   // N_PAD*256 elements
    int n = i >> 8, j = i & 255;
    float v = 0.0f;
    if (n < N_NODES_C && j < F_IN_C) v = x[n * F_IN_C + j];
    h[i] = v;
}

__global__ void zero4_kernel(float4* __restrict__ p, int n4) {
    int i = blockIdx.x * 256 + threadIdx.x;
    if (i < n4) p[i] = make_float4(0.f, 0.f, 0.f, 0.f);
}

// ---------------------------------------------------------------------------
// m = h @ Wg   ([N_PAD,256] x [256,256]) via bf16 WMMA, f32 accumulate.
// Block: 256 threads = 8 waves; waves 0-3 rows [r0,r0+16) cols w*64,
// waves 4-7 rows [r0+16,r0+32).  4 accumulators per wave, K-loop 8 x 32.
__global__ void __launch_bounds__(256)
gemm_m_kernel(const float* __restrict__ h, const __bf16* __restrict__ WgT,
              float* __restrict__ m) {
    int lane = threadIdx.x & 31;
    int wave = threadIdx.x >> 5;
    int half = lane >> 4;
    int row0 = blockIdx.x * 32 + (wave >> 2) * 16;
    int c0   = (wave & 3) * 64;
    int rowA = row0 + (lane & 15);

    v8f acc0 = {}, acc1 = {}, acc2 = {}, acc3 = {};
#pragma unroll
    for (int k0 = 0; k0 < 256; k0 += 32) {
        v16bf a = load_a_frag_f32(h, rowA, 256, k0, lane);
        v16bf b0 = load_b_frag_bf16(WgT, c0 +  0 + (lane & 15), 256, k0, lane);
        acc0 = wmma_bf16(a, b0, acc0);
        v16bf b1 = load_b_frag_bf16(WgT, c0 + 16 + (lane & 15), 256, k0, lane);
        acc1 = wmma_bf16(a, b1, acc1);
        v16bf b2 = load_b_frag_bf16(WgT, c0 + 32 + (lane & 15), 256, k0, lane);
        acc2 = wmma_bf16(a, b2, acc2);
        v16bf b3 = load_b_frag_bf16(WgT, c0 + 48 + (lane & 15), 256, k0, lane);
        acc3 = wmma_bf16(a, b3, acc3);
    }
    // C/D layout: VGPR r -> row = r + 8*half, col = lane&15
    v8f accs[4] = {acc0, acc1, acc2, acc3};
#pragma unroll
    for (int t = 0; t < 4; ++t) {
        int c = c0 + t * 16 + (lane & 15);
#pragma unroll
        for (int vr = 0; vr < 8; ++vr)
            m[(long long)(row0 + vr + 8 * half) * 256 + c] = accs[t][vr];
    }
}

// ---------------------------------------------------------------------------
// agg[dst] += m[src] * w   — thread per (edge, 4-feature chunk).
// m/agg are L2-resident (51 MB each << 192 MB), atomics resolve in L2.
__global__ void __launch_bounds__(256)
scatter_kernel(const float* __restrict__ m, const int* __restrict__ ei,
               const float* __restrict__ ew, float* __restrict__ agg) {
    int tid = blockIdx.x * 256 + threadIdx.x;      // E * 64 threads
    int e = tid >> 6;
    if (e >= N_EDGES_C) return;
    int f = (tid & 63) << 2;
    int s = ei[e];
    int d = ei[N_EDGES_C + e];
    float w = ew[e];
    float4 v = *(const float4*)(m + (long long)s * 256 + f);
    float* dst = agg + (long long)d * 256 + f;
    atomicAdd(dst + 0, v.x * w);
    atomicAdd(dst + 1, v.y * w);
    atomicAdd(dst + 2, v.z * w);
    atomicAdd(dst + 3, v.w * w);
}

// ---------------------------------------------------------------------------
// Fused GRU cell: h = GRU(agg, h).  Block = 16 rows x 256 cols, 8 waves;
// wave owns 32 output columns -> 12 WMMA accumulators:
//   acc[mat][gate][tile], mat 0 = agg@w_ih^T, mat 1 = h@w_hh^T.
// In-place h update (barrier separates all h reads from writes; each
// (row,col) is owned by exactly one thread).
__global__ void __launch_bounds__(256)
gru_kernel(const float* __restrict__ agg, float* __restrict__ h,
           const __bf16* __restrict__ wihb, const __bf16* __restrict__ whhb,
           const float* __restrict__ bih, const float* __restrict__ bhh) {
    int lane = threadIdx.x & 31;
    int wave = threadIdx.x >> 5;
    int half = lane >> 4;
    int row0 = blockIdx.x * 16;
    int c0   = wave * 32;
    int rowA = row0 + (lane & 15);

    v8f acc[2][3][2];
#pragma unroll
    for (int a = 0; a < 2; ++a)
#pragma unroll
        for (int g = 0; g < 3; ++g)
#pragma unroll
            for (int t = 0; t < 2; ++t) { v8f z = {}; acc[a][g][t] = z; }

#pragma unroll
    for (int k0 = 0; k0 < 256; k0 += 32) {
        v16bf a_agg = load_a_frag_f32(agg, rowA, 256, k0, lane);
        v16bf a_h   = load_a_frag_f32(h,   rowA, 256, k0, lane);
#pragma unroll
        for (int g = 0; g < 3; ++g) {
#pragma unroll
            for (int t = 0; t < 2; ++t) {
                int col = g * 256 + c0 + t * 16 + (lane & 15);
                v16bf b_ih = load_b_frag_bf16(wihb, col, 256, k0, lane);
                acc[0][g][t] = wmma_bf16(a_agg, b_ih, acc[0][g][t]);
                v16bf b_hh = load_b_frag_bf16(whhb, col, 256, k0, lane);
                acc[1][g][t] = wmma_bf16(a_h, b_hh, acc[1][g][t]);
            }
        }
    }

    __syncthreads();   // all waves done reading h rows before in-place update

#pragma unroll
    for (int t = 0; t < 2; ++t) {
        int c = c0 + t * 16 + (lane & 15);
        float bir = bih[c],       bhr = bhh[c];
        float biz = bih[256 + c], bhz = bhh[256 + c];
        float bin = bih[512 + c], bhn = bhh[512 + c];
#pragma unroll
        for (int vr = 0; vr < 8; ++vr) {
            long long idx = (long long)(row0 + vr + 8 * half) * 256 + c;
            float hold = h[idx];
            float r  = sigmoidf_(acc[0][0][t][vr] + bir + acc[1][0][t][vr] + bhr);
            float z  = sigmoidf_(acc[0][1][t][vr] + biz + acc[1][1][t][vr] + bhz);
            float nn = tanhf(acc[0][2][t][vr] + bin + r * (acc[1][2][t][vr] + bhn));
            h[idx] = (1.0f - z) * nn + z * hold;
        }
    }
}

// ---------------------------------------------------------------------------
// out = softmax(relu(h) @ fc_w^T + fc_b); fc_w staged in LDS (8 KB).
__global__ void __launch_bounds__(256)
final_kernel(const float* __restrict__ h, const float* __restrict__ fcw,
             const float* __restrict__ fcb, float* __restrict__ out) {
    __shared__ float sw[N_ACT_C * 256];
    __shared__ float sb[N_ACT_C];
    for (int i = threadIdx.x; i < N_ACT_C * 256; i += 256) sw[i] = fcw[i];
    if (threadIdx.x < N_ACT_C) sb[threadIdx.x] = fcb[threadIdx.x];
    __syncthreads();

    int n = blockIdx.x * 256 + threadIdx.x;
    if (n >= N_NODES_C) return;

    float acc[N_ACT_C];
#pragma unroll
    for (int a = 0; a < N_ACT_C; ++a) acc[a] = sb[a];
    const float* hr = h + (long long)n * 256;
    for (int k = 0; k < 256; ++k) {
        float hv = fmaxf(hr[k], 0.0f);
#pragma unroll
        for (int a = 0; a < N_ACT_C; ++a) acc[a] += hv * sw[a * 256 + k];
    }
    float mx = acc[0];
#pragma unroll
    for (int a = 1; a < N_ACT_C; ++a) mx = fmaxf(mx, acc[a]);
    float s = 0.0f;
#pragma unroll
    for (int a = 0; a < N_ACT_C; ++a) { acc[a] = __expf(acc[a] - mx); s += acc[a]; }
    float inv = 1.0f / s;
#pragma unroll
    for (int a = 0; a < N_ACT_C; ++a) out[(long long)n * N_ACT_C + a] = acc[a] * inv;
}

// ---------------------------------------------------------------------------
extern "C" void kernel_launch(void* const* d_in, const int* in_sizes, int n_in,
                              void* d_out, int out_size, void* d_ws, size_t ws_size,
                              hipStream_t stream) {
    const float* x   = (const float*)d_in[0];
    const int*   ei  = (const int*)  d_in[1];
    const float* ew  = (const float*)d_in[2];
    const float* gg  = (const float*)d_in[3];
    const float* wih = (const float*)d_in[4];
    const float* whh = (const float*)d_in[5];
    const float* bih = (const float*)d_in[6];
    const float* bhh = (const float*)d_in[7];
    const float* fcw = (const float*)d_in[8];
    const float* fcb = (const float*)d_in[9];
    float* out = (float*)d_out;

    // Workspace layout (all 1 KB aligned):
    //   h, m, agg : N_PAD*256 f32 each (51.2 MB each)
    //   ggT (3x256x256), wihb (768x256), whhb (768x256) bf16
    float* h   = (float*)d_ws;
    float* m   = h + (size_t)N_PAD * 256;
    float* agg = m + (size_t)N_PAD * 256;
    __bf16* ggT  = (__bf16*)(agg + (size_t)N_PAD * 256);
    __bf16* wihb = ggT + 3 * 256 * 256;
    __bf16* whhb = wihb + 768 * 256;

    convw_kernel<<<(3 * 256 * 256 + 255) / 256, 256, 0, stream>>>(gg, wih, whh,
                                                                  ggT, wihb, whhb);
    init_h_kernel<<<N_PAD, 256, 0, stream>>>(x, h);

    for (int l = 0; l < 3; ++l) {
        gemm_m_kernel<<<N_PAD / 32, 256, 0, stream>>>(h, ggT + (size_t)l * 65536, m);
        zero4_kernel<<<(N_PAD * 64 + 255) / 256, 256, 0, stream>>>((float4*)agg,
                                                                   N_PAD * 64);
        scatter_kernel<<<(N_EDGES_C * 64) / 256, 256, 0, stream>>>(m, ei, ew, agg);
        gru_kernel<<<N_PAD / 16, 256, 0, stream>>>(agg, h, wihb, whhb, bih, bhh);
    }

    final_kernel<<<(N_NODES_C + 255) / 256, 256, 0, stream>>>(h, fcw, fcb, out);
}
